// TRDeepFeatureMatcher_46548855554442
// MI455X (gfx1250) — matrix operand order repaired and checked by reading, not compiled, not run on hardware
//
#include <hip/hip_runtime.h>

typedef __attribute__((ext_vector_type(16))) __bf16 v16bf;
typedef __attribute__((ext_vector_type(8)))  __bf16 v8bf;
typedef __attribute__((ext_vector_type(8)))  float  v8f;

#define N_PTS    9216      // 96*96
#define C_DESC   256
#define W_MAP    96
#define W_ACT    192
#define HW_ACT   36864     // 192*192
#define N_TILES  576       // N_PTS/16
#define K_CHUNKS 8         // 256/32

// output layout (floats), reference tuple flattened in return order
#define OFF_A 46080        // out_float = N*5
#define OFF_B 119808       // + refined_A = 2*N*4
#define OFF_M 193536       // + refined_B = 2*N*4 ; total 230400

__device__ __forceinline__ unsigned short f2bf(float f) {
  unsigned int u = __float_as_uint(f);
  u += 0x7FFFu + ((u >> 16) & 1u);          // round-to-nearest-even
  return (unsigned short)(u >> 16);
}

// ---- per-pixel L2 normalize map (C,N) -> bf16 descriptors, row-major (N,C) ----
__global__ void norm_desc_kernel(const float* __restrict__ map,
                                 unsigned short* __restrict__ out) {
  int n = blockIdx.x * blockDim.x + threadIdx.x;
  float s = 0.f;
  for (int c = 0; c < C_DESC; ++c) { float v = map[c * N_PTS + n]; s += v * v; }
  float r = rsqrtf(s);
  for (int c = 0; c < C_DESC; ++c) out[n * C_DESC + c] = f2bf(map[c * N_PTS + n] * r);
}

// ---- per-position inverse L2 norm of activations (C,H2,W2) ----
__global__ void rnorm_act_kernel(const float* __restrict__ act,
                                 float* __restrict__ rn) {
  int p = blockIdx.x * blockDim.x + threadIdx.x;
  float s = 0.f;
  for (int c = 0; c < C_DESC; ++c) { float v = act[c * HW_ACT + p]; s += v * v; }
  rn[p] = rsqrtf(s);
}

// B fragment loader: col = lo16, lane holds K = hi*8..+7 and 16+hi*8..+7.
__device__ __forceinline__ void load_btile(const unsigned short* __restrict__ bbase,
                                           int j, v16bf* bf) {
  const unsigned short* bcol = bbase + (size_t)j * (16 * C_DESC);
  #pragma unroll
  for (int kc = 0; kc < K_CHUNKS; ++kc) {
    v8bf l0 = *(const v8bf*)(bcol + kc * 32);
    v8bf l1 = *(const v8bf*)(bcol + kc * 32 + 16);
    v16bf f;
    #pragma unroll
    for (int e = 0; e < 8; ++e) { f[e] = l0[e]; f[e + 8] = l1[e]; }
    bf[kc] = f;
  }
}

// ---- fused NT-GEMM (A @ B^T, both (N,C) bf16) with per-row top-2 reduction ----
__global__ void __launch_bounds__(256)
gemm_top2_kernel(const unsigned short* __restrict__ A,
                 const unsigned short* __restrict__ B,
                 float* __restrict__ s1o, float* __restrict__ s2o,
                 int* __restrict__ i1o) {
  const int wave = threadIdx.x >> 5;
  const int lane = threadIdx.x & 31;
  const int hi   = lane >> 4;
  const int lo16 = lane & 15;
  const int m0   = (blockIdx.x * 8 + wave) * 16;

  // Resident A fragments: 16x32 bf16 per K-chunk.
  // Lane layout: row = lo16, holds K = hi*8..+7 (v0-3) and 16+hi*8..+7 (v4-7).
  v16bf afrag[K_CHUNKS];
  {
    const unsigned short* arow = A + (size_t)(m0 + lo16) * C_DESC + hi * 8;
    #pragma unroll
    for (int kc = 0; kc < K_CHUNKS; ++kc) {
      v8bf l0 = *(const v8bf*)(arow + kc * 32);
      v8bf l1 = *(const v8bf*)(arow + kc * 32 + 16);
      v16bf f;
      #pragma unroll
      for (int e = 0; e < 8; ++e) { f[e] = l0[e]; f[e + 8] = l1[e]; }
      afrag[kc] = f;
    }
  }

  float t1[8], t2[8]; int ix[8];
  #pragma unroll
  for (int v = 0; v < 8; ++v) { t1[v] = -3.0e38f; t2[v] = -3.0e38f; ix[v] = 0; }

  // branchless top-2 insert (v_cmp + v_cndmask, no EXEC churn around WMMA)
  auto update = [&](const v8f& acc, int ncol) {
    #pragma unroll
    for (int v = 0; v < 8; ++v) {
      float s = acc[v];                    // row m0 + v + 8*hi, col ncol
      bool g = s > t1[v];
      t2[v] = g ? t1[v] : fmaxf(t2[v], s);
      ix[v] = g ? ncol : ix[v];
      t1[v] = g ? s : t1[v];
    }
  };

  const unsigned short* bbase = B + (size_t)lo16 * C_DESC + hi * 8;
  v16bf b0[K_CHUNKS], b1[K_CHUNKS];
  load_btile(bbase, 0, b0);

  // software pipeline: prefetch next N-tile while WMMAs consume current one
  for (int j = 0; j < N_TILES; j += 2) {
    load_btile(bbase, j + 1, b1);
    {
      v8f acc = {};
      #pragma unroll
      for (int kc = 0; kc < K_CHUNKS; ++kc)
        acc = __builtin_amdgcn_wmma_f32_16x16x32_bf16(
            false, afrag[kc], false, b0[kc], (short)0, acc, false, false);
      update(acc, j * 16 + lo16);
    }
    load_btile(bbase, (j + 2 < N_TILES) ? j + 2 : 0, b0);
    {
      v8f acc = {};
      #pragma unroll
      for (int kc = 0; kc < K_CHUNKS; ++kc)
        acc = __builtin_amdgcn_wmma_f32_16x16x32_bf16(
            false, afrag[kc], false, b1[kc], (short)0, acc, false, false);
      update(acc, (j + 1) * 16 + lo16);
    }
  }

  // merge top-2 sets across the 16 lanes that share each output row (branchless)
  #pragma unroll
  for (int v = 0; v < 8; ++v) {
    float a1 = t1[v], a2 = t2[v]; int ii = ix[v];
    #pragma unroll
    for (int m = 1; m <= 8; m <<= 1) {
      float o1 = __shfl_xor(a1, m, 32);
      float o2 = __shfl_xor(a2, m, 32);
      int   oi = __shfl_xor(ii, m, 32);
      bool g = o1 > a1;
      float n2 = g ? fmaxf(a1, o2) : fmaxf(a2, o1);
      a1 = g ? o1 : a1;
      ii = g ? oi : ii;
      a2 = n2;
    }
    if (lo16 == 0) {
      int row = m0 + v + hi * 8;
      s1o[row] = a1; s2o[row] = a2; i1o[row] = ii;
    }
  }
}

// ---- mutual-NN + ratio + border mask; also writes match_sim and refined_A ----
__global__ void valid_kernel(const float* __restrict__ s12a, const float* __restrict__ s12b,
                             const int* __restrict__ i12,
                             const float* __restrict__ s21a, const float* __restrict__ s21b,
                             const int* __restrict__ i21,
                             int* __restrict__ validArr,
                             int* __restrict__ pax, int* __restrict__ pay,
                             int* __restrict__ pbx, int* __restrict__ pby,
                             float* __restrict__ out) {
  int n = blockIdx.x * blockDim.x + threadIdx.x;
  float d0 = 2.f - 2.f * s12a[n];
  float d1v = 2.f - 2.f * s12b[n];
  float r12 = d0 / (d1v + 1e-8f);
  int nn = i12[n];
  float e0 = 2.f - 2.f * s21a[nn];
  float e1 = 2.f - 2.f * s21b[nn];
  float r21 = e0 / (e1 + 1e-8f);
  bool mnn = (i21[nn] == n) && (r12 <= 0.95f) && (r21 <= 0.95f);
  int xA = n % W_MAP, yA = n / W_MAP, xB = nn % W_MAP, yB = nn / W_MAP;
  bool disc = (xA == 0) || (xA == W_MAP - 1) || (yA == 0) || (yA == W_MAP - 1) ||
              (xB == 0) || (xB == W_MAP - 1) || (yB == 0) || (yB == W_MAP - 1);
  bool val = mnn && !disc;
  int ax = (val ? xA : 1) * 2, ay = (val ? yA : 1) * 2;
  int bx = (val ? xB : 1) * 2, by = (val ? yB : 1) * 2;
  validArr[n] = val ? 1 : 0;
  pax[n] = ax; pay[n] = ay; pbx[n] = bx; pby[n] = by;
  out[n * 5] = s12a[n];
  const int nx[4] = {0, 0, 1, 1}, ny[4] = {0, 1, 0, 1};
  #pragma unroll
  for (int i = 0; i < 4; ++i) {
    out[OFF_A + n * 4 + i]             = (float)(ax + nx[i]);
    out[OFF_A + N_PTS * 4 + n * 4 + i] = (float)(ay + ny[i]);
  }
}

// ---- one wave per match: 4x4 scores, ratio tests, cycle check, discard top-2 ----
__global__ void __launch_bounds__(256)
refine_kernel(const float* __restrict__ actA, const float* __restrict__ actB,
              const float* __restrict__ rnA, const float* __restrict__ rnB,
              const int* __restrict__ pax, const int* __restrict__ pay,
              const int* __restrict__ pbx, const int* __restrict__ pby,
              const int* __restrict__ validArr,
              float* __restrict__ out) {
  const int n    = blockIdx.x * 8 + (threadIdx.x >> 5);
  const int lane = threadIdx.x & 31;
  const int pair = lane & 15;
  const int ti = pair >> 2, tj = pair & 3;
  const int ch0 = (lane >> 4) * 128;
  const int nx[4] = {0, 0, 1, 1}, ny[4] = {0, 1, 0, 1};

  int ax = pax[n], ay = pay[n], bx = pbx[n], by = pby[n];
  int posA = (ay + ny[ti]) * W_ACT + (ax + nx[ti]);
  int posB = (by + ny[tj]) * W_ACT + (bx + nx[tj]);
  const float* pa = actA + posA + (size_t)ch0 * HW_ACT;
  const float* pb = actB + posB + (size_t)ch0 * HW_ACT;
  float s = 0.f;
  #pragma unroll 4
  for (int c = 0; c < 128; ++c) s += pa[(size_t)c * HW_ACT] * pb[(size_t)c * HW_ACT];
  s += __shfl_xor(s, 16, 32);
  s *= rnA[posA] * rnB[posB];

  float S[16];
  #pragma unroll
  for (int p = 0; p < 16; ++p) S[p] = __shfl(s, p, 32);

  float ratioA[4], scoreA[4]; int matchA[4];
  #pragma unroll
  for (int i = 0; i < 4; ++i) {
    float a1 = -3.0e38f, a2 = -3.0e38f; int ja = 0;
    #pragma unroll
    for (int j = 0; j < 4; ++j) {
      float v = S[i * 4 + j];
      bool g = v > a1;
      a2 = g ? a1 : fmaxf(a2, v);
      ja = g ? j : ja;
      a1 = g ? v : a1;
    }
    float dd0 = 2.f - 2.f * a1, dd1 = 2.f - 2.f * a2;
    ratioA[i] = dd0 / (dd1 + 1e-8f);
    matchA[i] = ja; scoreA[i] = dd0;
  }
  float ratioB[4]; int matchB[4];
  #pragma unroll
  for (int j = 0; j < 4; ++j) {
    float b1 = -3.0e38f, b2 = -3.0e38f; int ib = 0;
    #pragma unroll
    for (int i = 0; i < 4; ++i) {
      float v = S[i * 4 + j];
      bool g = v > b1;
      b2 = g ? b1 : fmaxf(b2, v);
      ib = g ? i : ib;
      b1 = g ? v : b1;
    }
    float dd0 = 2.f - 2.f * b1, dd1 = 2.f - 2.f * b2;
    ratioB[j] = dd0 / (dd1 + 1e-8f);
    matchB[j] = ib;
  }
  bool rm[4]; float sm[4];
  #pragma unroll
  for (int i = 0; i < 4; ++i) {
    bool cyc = (matchB[matchA[i]] == i);
    rm[i] = (fmaxf(ratioA[i], ratioB[i]) < 0.9f) && cyc;
    sm[i] = rm[i] ? scoreA[i] : 5.0f;
  }
  int d0 = 0; float v0 = sm[0];
  #pragma unroll
  for (int i = 1; i < 4; ++i) { bool g = sm[i] > v0; v0 = g ? sm[i] : v0; d0 = g ? i : d0; }
  int d1i = (d0 == 0) ? 1 : 0; float v1 = sm[d1i];
  #pragma unroll
  for (int i = 0; i < 4; ++i) {
    bool g = (i != d0) && (sm[i] > v1);
    v1 = g ? sm[i] : v1; d1i = g ? i : d1i;
  }
  rm[d0] = false; rm[d1i] = false;
  bool val = validArr[n] != 0;

  if (lane == 0) {
    #pragma unroll
    for (int i = 0; i < 4; ++i) {
      out[n * 5 + 1 + i] = sm[i];
      out[OFF_B + n * 4 + i]             = (float)(bx + nx[matchA[i]]);
      out[OFF_B + N_PTS * 4 + n * 4 + i] = (float)(by + ny[matchA[i]]);
      out[OFF_M + n * 4 + i] = (rm[i] && val) ? 1.f : 0.f;
    }
  }
}

extern "C" void kernel_launch(void* const* d_in, const int* in_sizes, int n_in,
                              void* d_out, int out_size, void* d_ws, size_t ws_size,
                              hipStream_t stream) {
  const float* mapA = (const float*)d_in[0];
  const float* mapB = (const float*)d_in[1];
  const float* actA = (const float*)d_in[2];
  const float* actB = (const float*)d_in[3];
  float* out = (float*)d_out;
  char* ws = (char*)d_ws;

  size_t off = 0;
  auto take = [&](size_t bytes) {
    size_t o = off; off += (bytes + 255) & ~(size_t)255; return o;
  };
  unsigned short* d1 = (unsigned short*)(ws + take((size_t)N_PTS * C_DESC * 2));
  unsigned short* d2 = (unsigned short*)(ws + take((size_t)N_PTS * C_DESC * 2));
  float* s12a = (float*)(ws + take(N_PTS * 4));
  float* s12b = (float*)(ws + take(N_PTS * 4));
  int*   i12  = (int*)  (ws + take(N_PTS * 4));
  float* s21a = (float*)(ws + take(N_PTS * 4));
  float* s21b = (float*)(ws + take(N_PTS * 4));
  int*   i21  = (int*)  (ws + take(N_PTS * 4));
  float* rnA  = (float*)(ws + take(HW_ACT * 4));
  float* rnB  = (float*)(ws + take(HW_ACT * 4));
  int* validArr = (int*)(ws + take(N_PTS * 4));
  int* pax = (int*)(ws + take(N_PTS * 4));
  int* pay = (int*)(ws + take(N_PTS * 4));
  int* pbx = (int*)(ws + take(N_PTS * 4));
  int* pby = (int*)(ws + take(N_PTS * 4));

  norm_desc_kernel<<<N_PTS / 256, 256, 0, stream>>>(mapA, d1);
  norm_desc_kernel<<<N_PTS / 256, 256, 0, stream>>>(mapB, d2);
  rnorm_act_kernel<<<HW_ACT / 256, 256, 0, stream>>>(actA, rnA);
  rnorm_act_kernel<<<HW_ACT / 256, 256, 0, stream>>>(actB, rnB);

  // row-direction top-2 of sim, then column-direction via swapped operands
  gemm_top2_kernel<<<N_TILES / 8, 256, 0, stream>>>(d1, d2, s12a, s12b, i12);
  gemm_top2_kernel<<<N_TILES / 8, 256, 0, stream>>>(d2, d1, s21a, s21b, i21);

  valid_kernel<<<N_PTS / 256, 256, 0, stream>>>(s12a, s12b, i12, s21a, s21b, i21,
                                                validArr, pax, pay, pbx, pby, out);
  refine_kernel<<<N_PTS / 8, 256, 0, stream>>>(actA, actB, rnA, rnB,
                                               pax, pay, pbx, pby, validArr, out);
}
